// QMeasureDensity_83047487635527
// MI455X (gfx1250) — matrix operand order repaired and checked by reading, not compiled, not run on hardware
//
#include <hip/hip_runtime.h>

#define BATCH   8192
#define DIM     4096
#define NK      (DIM / 32)     // 128 k-chunks
#define WAVES   4              // waves per workgroup (128 threads)
#define MT      8              // 16-wide m-tiles processed per pass (128 columns)
#define KC      32             // k-chunk per WMMA step (bf16 K=32)
#define MSPLIT  2              // split of m-range across grid.y
#define MHALF   (DIM / MSPLIT)             // 2048
#define MGROUPS (MHALF / (MT * 16))        // 16
#define SLOT_BYTES 48                       // per-lane fragment slot (bank spread, 16B aligned)
#define LDS_ARR (MT * 32 * SLOT_BYTES)      // 12 KB per (hi|lo) array
#define TILE_ELEMS 4096                     // bf16 elems per (m-group, k-chunk) tile

typedef __attribute__((ext_vector_type(16))) __bf16 v16bf;
typedef __attribute__((ext_vector_type(8)))  __bf16 v8bf;
typedef __attribute__((ext_vector_type(8)))  float  v8f;
typedef __attribute__((ext_vector_type(4)))  float  v4f;
typedef __attribute__((ext_vector_type(4)))  int    v4i;

typedef __attribute__((address_space(1))) v4i* g_v4i_p;   // global int4*
typedef __attribute__((address_space(3))) v4i* l_v4i_p;   // LDS int4*

#if defined(__gfx1250__) && __has_builtin(__builtin_amdgcn_global_load_async_to_lds_b128)
#define HAVE_ASYNC 1
#else
#define HAVE_ASYNC 0
#endif

#if HAVE_ASYNC
#if __has_builtin(__builtin_amdgcn_s_wait_asynccnt)
#define WAIT_ASYNC(N) __builtin_amdgcn_s_wait_asynccnt(N)
#else
#define WAIT_ASYNC(N) asm volatile("s_wait_asynccnt %0" :: "i"(N) : "memory")
#endif
#else
#define WAIT_ASYNC(N) do {} while (0)
#endif

// Copy one 8KB hi tile + 8KB lo tile (pre-swizzled bf16 fragments) into LDS.
// chunk c (16B) -> slot c>>1, half c&1; LDS slot stride 48B for bank spread.
__device__ __forceinline__ void stage_tile(const __bf16* gH, const __bf16* gL,
                                           char* ldsBase, int tid)
{
#pragma unroll
    for (int i = 0; i < 4; ++i) {
        int c   = i * (WAVES * 32) + tid;                 // 0..511
        int dst = (c >> 1) * SLOT_BYTES + (c & 1) * 16;
#if HAVE_ASYNC
        __builtin_amdgcn_global_load_async_to_lds_b128(
            (g_v4i_p)(gH + c * 8),
            (l_v4i_p)(ldsBase + dst), 0, 0);
        __builtin_amdgcn_global_load_async_to_lds_b128(
            (g_v4i_p)(gL + c * 8),
            (l_v4i_p)(ldsBase + LDS_ARR + dst), 0, 0);
#else
        *(v8bf*)(ldsBase + dst)           = *(const v8bf*)(gH + c * 8);
        *(v8bf*)(ldsBase + LDS_ARR + dst) = *(const v8bf*)(gL + c * 8);
#endif
    }
}

// One-time rho f32 -> bf16 hi/lo split, stored in WMMA-B fragment-swizzled order:
// tile = (m/128)*NK + (k/32); within tile: slot = mt*32 + (kk/16)*16 + n, elem = kk&15.
__global__ __launch_bounds__(256)
void qmd_prep(const float* __restrict__ rho, __bf16* __restrict__ rhoH,
              __bf16* __restrict__ rhoL)
{
    size_t idx = (size_t)blockIdx.x * 256 + threadIdx.x;   // over DIM*DIM
    int m = (int)(idx & (DIM - 1));
    int k = (int)(idx >> 12);
    float g  = rho[idx];
    __bf16 h = (__bf16)g;
    __bf16 l = (__bf16)(g - (float)h);
    int j  = m & 127, kk = k & 31;
    int mt = j >> 4, n = j & 15;
    int slot = mt * 32 + ((kk >> 4) * 16) + n;
    size_t tile = (size_t)(m >> 7) * NK + (size_t)(k >> 5);
    size_t off  = tile * TILE_ELEMS + (size_t)slot * 16 + (kk & 15);
    rhoH[off] = h;
    rhoL[off] = l;
}

__global__ __launch_bounds__(WAVES * 32)
void qmd_main(const float* __restrict__ X,
              const __bf16* __restrict__ rhoH, const __bf16* __restrict__ rhoL,
              float* __restrict__ wsq, float* __restrict__ wsn)
{
    __shared__ char lds[2 * 2 * LDS_ARR];  // double-buffered (hi|lo) fragment arrays

    const int tid  = threadIdx.x;
    const int wave = tid >> 5;
    const int lane = tid & 31;
    const int hf   = lane >> 4;            // half-wave id (0/1)
    const int l16  = lane & 15;
    const int b0   = (blockIdx.x * WAVES + wave) * 16;

    float pq[8], pn[8];
#pragma unroll
    for (int r = 0; r < 8; ++r) { pq[r] = 0.0f; pn[r] = 0.0f; }

    const float* xrow = X + (size_t)(b0 + l16) * DIM;  // row for this lane's A fragment

    for (int mg = 0; mg < MGROUPS; ++mg) {
        const int    mgAbs    = blockIdx.y * MGROUPS + mg;
        const int    m0       = mgAbs * (MT * 16);
        const size_t tileBase = (size_t)mgAbs * NK;

        v8f c[MT];
#pragma unroll
        for (int t = 0; t < MT; ++t)
#pragma unroll
            for (int i = 0; i < 8; ++i) c[t][i] = 0.0f;

        // Prologue: stage k-chunk 0 into buffer 0.
        stage_tile(rhoH + tileBase * TILE_ELEMS, rhoL + tileBase * TILE_ELEMS,
                   lds, tid);

        for (int kc = 0; kc < NK; ++kc) {
            char* cur = lds + (kc & 1) * (2 * LDS_ARR);
            char* nxt = lds + ((kc + 1) & 1) * (2 * LDS_ARR);
            if (kc + 1 < NK) {   // prefetch next chunk, then wait for current
                size_t tg = (tileBase + kc + 1) * TILE_ELEMS;
                stage_tile(rhoH + tg, rhoL + tg, nxt, tid);
                WAIT_ASYNC(8);
            } else {
                WAIT_ASYNC(0);
            }
            __syncthreads();     // current buffer fully written by all waves

            const int k0 = kc * KC;
            // A fragment (contiguous k order, matching B): elems 0..15 -> K=k0+16*hf+e.
            // Four coalesced b128 loads.
            v4f a0 = *(const v4f*)(xrow + k0 + 16 * hf);
            v4f a1 = *(const v4f*)(xrow + k0 + 16 * hf + 4);
            v4f a2 = *(const v4f*)(xrow + k0 + 16 * hf + 8);
            v4f a3 = *(const v4f*)(xrow + k0 + 16 * hf + 12);
            float af[16];
#pragma unroll
            for (int i = 0; i < 4; ++i) {
                af[i] = a0[i]; af[4 + i] = a1[i]; af[8 + i] = a2[i]; af[12 + i] = a3[i];
            }
            v16bf Ah, Al;
#pragma unroll
            for (int i = 0; i < 16; ++i) {
                __bf16 h = (__bf16)af[i];
                Ah[i] = h;
                Al[i] = (__bf16)(af[i] - (float)h);
            }

#pragma unroll
            for (int t = 0; t < MT; ++t) {
                const char* ph = cur + (t * 32 + lane) * SLOT_BYTES;
                const char* pl = ph + LDS_ARR;
                v8bf bh0 = *(const v8bf*)(ph);
                v8bf bh1 = *(const v8bf*)(ph + 16);
                v8bf bl0 = *(const v8bf*)(pl);
                v8bf bl1 = *(const v8bf*)(pl + 16);
                v16bf Bh, Bl;
#pragma unroll
                for (int i = 0; i < 8; ++i) {
                    Bh[i] = bh0[i]; Bh[8 + i] = bh1[i];
                    Bl[i] = bl0[i]; Bl[8 + i] = bl1[i];
                }
                // bf16x3 split-product GEMM: hi*hi + hi*lo + lo*hi (lo*lo < f32 eps)
                c[t] = __builtin_amdgcn_wmma_f32_16x16x32_bf16(
                           false, Ah, false, Bh, (short)0, c[t], false, false);
                c[t] = __builtin_amdgcn_wmma_f32_16x16x32_bf16(
                           false, Ah, false, Bl, (short)0, c[t], false, false);
                c[t] = __builtin_amdgcn_wmma_f32_16x16x32_bf16(
                           false, Al, false, Bh, (short)0, c[t], false, false);
            }
            __syncthreads();     // all waves done reading cur before it is restaged
        } // k loop

        // Combine Y tile with X: C element (r, lane) is (b = b0+r+8*hf, m = m0+t*16+l16).
        // The same X load feeds the norm^2 accumulation for free.
#pragma unroll
        for (int t = 0; t < MT; ++t) {
            int m = m0 + t * 16 + l16;
#pragma unroll
            for (int r = 0; r < 8; ++r) {
                float xv = X[(size_t)(b0 + 8 * hf + r) * DIM + m];
                pq[r] += c[t][r] * xv;
                pn[r] += xv * xv;
            }
        }
    } // m-group loop

    // Butterfly reduction across the 16 lanes of each half-wave.
#pragma unroll
    for (int r = 0; r < 8; ++r) {
        float q = pq[r], n = pn[r];
        for (int mask = 1; mask < 16; mask <<= 1) {
            q += __shfl_xor(q, mask, 32);
            n += __shfl_xor(n, mask, 32);
        }
        pq[r] = q; pn[r] = n;
    }
    if (l16 == 0) {
#pragma unroll
        for (int r = 0; r < 8; ++r) {
            int b = b0 + 8 * hf + r;
            wsq[(size_t)blockIdx.y * BATCH + b] = pq[r];
            wsn[(size_t)blockIdx.y * BATCH + b] = pn[r];
        }
    }
}

__global__ __launch_bounds__(256)
void qmd_final(const float* __restrict__ wsq, const float* __restrict__ wsn,
               float* __restrict__ out)
{
    int b = blockIdx.x * 256 + threadIdx.x;
    float q = wsq[b] + wsq[BATCH + b];
    float n = wsn[b] + wsn[BATCH + b];
    out[b] = n * q;
}

extern "C" void kernel_launch(void* const* d_in, const int* in_sizes, int n_in,
                              void* d_out, int out_size, void* d_ws, size_t ws_size,
                              hipStream_t stream)
{
    (void)in_sizes; (void)n_in; (void)out_size; (void)ws_size;
    const float* X   = (const float*)d_in[0];
    const float* rho = (const float*)d_in[1];
    float* out = (float*)d_out;

    float*  wsq  = (float*)d_ws;                              // [MSPLIT][BATCH]
    float*  wsn  = wsq + (size_t)MSPLIT * BATCH;              // [MSPLIT][BATCH]
    __bf16* rhoH = (__bf16*)((char*)d_ws + (size_t)2 * MSPLIT * BATCH * sizeof(float));
    __bf16* rhoL = rhoH + (size_t)DIM * DIM;                  // 32MB each

    qmd_prep<<<(DIM * (size_t)DIM) / 256, 256, 0, stream>>>(rho, rhoH, rhoL);

    dim3 grid(BATCH / (WAVES * 16), MSPLIT);                  // (128, 2)
    qmd_main<<<grid, WAVES * 32, 0, stream>>>(X, rhoH, rhoL, wsq, wsn);

    qmd_final<<<BATCH / 256, 256, 0, stream>>>(wsq, wsn, out);
}
